// ScaledDotProductAttention_57432302682805
// MI455X (gfx1250) — matrix operand order repaired and checked by reading, not compiled, not run on hardware
//
#include <hip/hip_runtime.h>

// ---------------------------------------------------------------------------
// SDPA forward for [B=2, H=16, S=2048, D=64] fp32, returning (out, attn).
// One workgroup (4 waves, 128 threads) per 16-row query tile.
//   Pass 1: scores = (Q/8) @ K^T via v_wmma_f32_16x16x32_bf16 -> LDS,
//           fusing the per-row running max into the WMMA epilogue.
//   Pass 2: row-max cross-lane reduce; exp(s-max) + row sum via float4 scans.
//   Pass 3: stream normalized attn (536 MB, the roofline term) with b128
//           stores and out = P @ V via WMMA, cross-wave reduce in LDS.
// ---------------------------------------------------------------------------

typedef __attribute__((ext_vector_type(16))) __bf16 v16bf;
typedef __attribute__((ext_vector_type(8)))  float  v8f;
typedef __attribute__((ext_vector_type(4)))  float  v4f;

static __device__ __forceinline__ __bf16 f2bf(float f) {
  return (__bf16)f;   // fptrunc (RNE) -> v_cvt_pk_bf16_f32 pairs
}

#define SEQ   2048
#define HDIM  64
#define QTILE 16
#define RM_STRIDE 65   // padded stride for 16x64 row-max partials (bank-safe)

extern "C" __global__ __launch_bounds__(128)
void sdpa_fwd_kernel(const float* __restrict__ q,
                     const float* __restrict__ k,
                     const float* __restrict__ v,
                     float* __restrict__ out,
                     float* __restrict__ attn)
{
  extern __shared__ float smem[];
  float* sc    = smem;                       // [16][2048] scores -> exp values
  float* red   = smem + QTILE * SEQ;         // [16][8] partial reductions
  float* stats = red + QTILE * 8;            // [0..15] rowmax, [16..31] 1/rowsum
  float* red2  = stats + 32;                 // [16][RM_STRIDE] row-max partials

  const int bh  = blockIdx.y;                // b*H + h
  const int q0  = blockIdx.x * QTILE;        // first query row of tile
  const int tid  = threadIdx.x;
  const int wave = tid >> 5;
  const int lane = tid & 31;
  const int lm   = lane & 15;                // m (A,C rows) or n (B,C cols)
  const int kg   = lane >> 4;                // lane-group (K-split)

  const float* qb = q + (size_t)bh * SEQ * HDIM;
  const float* kb = k + (size_t)bh * SEQ * HDIM;
  const float* vb = v + (size_t)bh * SEQ * HDIM;

  // ---- Load Q tile into WMMA A layout, two K-steps (d 0..31, 32..63),
  //      folding the 1/temperature (=1/8) scale in before bf16 conversion.
  v16bf qa0, qa1;
  {
    const float* qrow = qb + (size_t)(q0 + lm) * HDIM;
    const int base0 = kg * 16;       // step 0
    const int base1 = 32 + kg * 16;  // step 1
#pragma unroll
    for (int e = 0; e < 8; ++e)  qa0[e] = f2bf(qrow[base0 + e] * 0.125f);
#pragma unroll
    for (int e = 8; e < 16; ++e) qa0[e] = f2bf(qrow[base0 + e] * 0.125f);
#pragma unroll
    for (int e = 0; e < 8; ++e)  qa1[e] = f2bf(qrow[base1 + e] * 0.125f);
#pragma unroll
    for (int e = 8; e < 16; ++e) qa1[e] = f2bf(qrow[base1 + e] * 0.125f);
  }

  // ---- Pass 1: scores for this wave's 512-key chunk -> LDS, fused row max --
  const int key0w = wave * (SEQ / 4);        // 512 keys per wave
  float rmax[8];                             // running max, rows vv + 8*kg
#pragma unroll
  for (int vv = 0; vv < 8; ++vv) rmax[vv] = -3.0e38f;

  for (int t = 0; t < 32; ++t) {
    const int kbase = key0w + t * 16;
    // K tile in WMMA B layout: B[d][n] = K[kbase+n][d]; lane n=lm reads a
    // contiguous 16-float slice of its key row per K-step.
    const float* krow = kb + (size_t)(kbase + lm) * HDIM + kg * 16;
    v16bf b0, b1;
#pragma unroll
    for (int e = 0; e < 16; ++e) b0[e] = f2bf(krow[e]);
#pragma unroll
    for (int e = 0; e < 16; ++e) b1[e] = f2bf(krow[32 + e]);

    v8f c = {};
    c = __builtin_amdgcn_wmma_f32_16x16x32_bf16(false, qa0, false, b0,
                                                (short)0, c, false, false);
    c = __builtin_amdgcn_wmma_f32_16x16x32_bf16(false, qa1, false, b1,
                                                (short)0, c, false, false);
    // C layout: VGPR vv -> row vv + 8*kg, col = lm
#pragma unroll
    for (int vv = 0; vv < 8; ++vv) {
      const float s = c[vv];
      sc[(size_t)(vv + 8 * kg) * SEQ + kbase + lm] = s;
      rmax[vv] = (s > rmax[vv]) ? s : rmax[vv];
    }
  }
  // Publish per-lane row-max partials: row (vv+8kg), slot (wave*16+lm)
#pragma unroll
  for (int vv = 0; vv < 8; ++vv)
    red2[(vv + 8 * kg) * RM_STRIDE + wave * 16 + lm] = rmax[vv];
  __syncthreads();

  // ---- Pass 2a: reduce 64 row-max partials per row (2 stages) ----
  const int r   = tid >> 3;
  const int seg = tid & 7;
  {
    const float* p = red2 + (size_t)r * RM_STRIDE + seg * 8;
    float m = p[0];
#pragma unroll
    for (int j = 1; j < 8; ++j) { float x = p[j]; m = (x > m) ? x : m; }
    red[r * 8 + seg] = m;
  }
  __syncthreads();
  if (tid < 16) {
    float m = red[tid * 8];
#pragma unroll
    for (int i = 1; i < 8; ++i) {
      float x = red[tid * 8 + i];
      m = (x > m) ? x : m;
    }
    stats[tid] = m;
  }
  __syncthreads();

  // ---- Pass 2b: exp(s - max) in place + row sum (float4 scan) ----
  {
    const float m = stats[r];
    v4f* row = (v4f*)(sc + (size_t)r * SEQ + seg * 256);
    float ssum = 0.0f;
#pragma unroll 4
    for (int i = 0; i < 64; ++i) {
      v4f x = row[i];
      v4f e;
#pragma unroll
      for (int j = 0; j < 4; ++j) e[j] = __expf(x[j] - m);
      row[i] = e;
      ssum += (e[0] + e[1]) + (e[2] + e[3]);
    }
    red[r * 8 + seg] = ssum;
  }
  __syncthreads();
  if (tid < 16) {
    float ssum = 0.0f;
#pragma unroll
    for (int i = 0; i < 8; ++i) ssum += red[tid * 8 + i];
    stats[16 + tid] = 1.0f / ssum;
  }
  __syncthreads();

  // ---- Pass 3a: stream normalized attn to HBM (b128 stores, coalesced) ----
  {
    float* attn_tile = attn + ((size_t)bh * SEQ + q0) * SEQ;
    const v4f* scv = (const v4f*)sc;
#pragma unroll 4
    for (int i = 0; i < 64; ++i) {
      const int idx4 = tid + i * 128;        // linear over [16][512] float4s
      const int rr   = idx4 >> 9;            // row (2048/4 = 512 chunks/row)
      const int cc   = (idx4 & 511) * 4;     // starting column
      const float inv = stats[16 + rr];
      v4f x = scv[idx4];
#pragma unroll
      for (int j = 0; j < 4; ++j) x[j] *= inv;
      *(v4f*)(attn_tile + (size_t)rr * SEQ + cc) = x;
    }
  }

  // ---- Pass 3b: out = P @ V over this wave's 512 keys (16 tiles of K=32) --
  v8f acc0 = {}, acc1 = {}, acc2 = {}, acc3 = {};
  const float invm = stats[16 + lm];          // normalizer for A row m = lm
  for (int t = 0; t < 16; ++t) {
    const int kb2 = key0w + t * 32;
    // A = P[16 x 32] bf16, A layout: element e -> key offset
    const float* prow = sc + (size_t)lm * SEQ + kb2;
    v16bf pa;
#pragma unroll
    for (int e = 0; e < 8; ++e)  pa[e] = f2bf(prow[e + 16 * kg] * invm);
#pragma unroll
    for (int e = 8; e < 16; ++e) pa[e] = f2bf(prow[e + 16 * kg] * invm);

    // B = V[32 x 16] per 16-wide d-tile: B[kk][n] = V[kb2+kk][dt*16+n]
    const float* vbase = vb + (size_t)(kb2 + 16 * kg) * HDIM + lm;
#pragma unroll
    for (int dt = 0; dt < 4; ++dt) {
      const float* vcol = vbase + dt * 16;
      v16bf bv;
#pragma unroll
      for (int e = 0; e < 16; ++e) bv[e] = f2bf(vcol[(size_t)e * HDIM]);
      v8f* accp = (dt == 0) ? &acc0 : (dt == 1) ? &acc1 : (dt == 2) ? &acc2 : &acc3;
      *accp = __builtin_amdgcn_wmma_f32_16x16x32_bf16(false, pa, false, bv,
                                                      (short)0, *accp, false, false);
    }
  }

  __syncthreads();   // score region free now; reuse for per-wave partials

  // Store wave partials: [wave][16][64] f32 (4 KB per wave)
  {
    float* part = smem + (size_t)wave * (QTILE * HDIM);
    v8f accs[4] = {acc0, acc1, acc2, acc3};
#pragma unroll
    for (int dt = 0; dt < 4; ++dt)
#pragma unroll
      for (int vv = 0; vv < 8; ++vv)
        part[(vv + 8 * kg) * HDIM + dt * 16 + lm] = accs[dt][vv];
  }
  __syncthreads();

  // Reduce 4 wave partials and write out tile (16x64 = 1024 elems, float4) --
  {
    float* out_tile = out + ((size_t)bh * SEQ + q0) * HDIM;
    const v4f* p0 = (const v4f*)smem;
    const v4f* p1 = (const v4f*)(smem + QTILE * HDIM);
    const v4f* p2 = (const v4f*)(smem + 2 * QTILE * HDIM);
    const v4f* p3 = (const v4f*)(smem + 3 * QTILE * HDIM);
    for (int i4 = tid; i4 < (QTILE * HDIM) / 4; i4 += 128) {
      v4f s = p0[i4];
      v4f a = p1[i4], b = p2[i4], c = p3[i4];
#pragma unroll
      for (int j = 0; j < 4; ++j) s[j] = (s[j] + a[j]) + (b[j] + c[j]);
      *(v4f*)(out_tile + (size_t)i4 * 4) = s;
    }
  }
}

extern "C" void kernel_launch(void* const* d_in, const int* in_sizes, int n_in,
                              void* d_out, int out_size, void* d_ws, size_t ws_size,
                              hipStream_t stream) {
  (void)in_sizes; (void)n_in; (void)out_size; (void)d_ws; (void)ws_size;
  const float* q = (const float*)d_in[0];
  const float* k = (const float*)d_in[1];
  const float* v = (const float*)d_in[2];

  const int B = 2, H = 16;
  float* out  = (float*)d_out;                              // [B,H,S,D]
  float* attn = out + (size_t)B * H * SEQ * HDIM;           // [B,H,S,S]

  // LDS: scores + red + stats + padded row-max partials
  const size_t shmem =
      (size_t)(QTILE * SEQ + QTILE * 8 + 32 + QTILE * RM_STRIDE) * sizeof(float);

  dim3 grid(SEQ / QTILE, B * H);   // (128, 32)
  dim3 block(128);                 // 4 waves of 32
  sdpa_fwd_kernel<<<grid, block, shmem, stream>>>(q, k, v, out, attn);
}
